// GT_87625922773239
// MI455X (gfx1250) — matrix operand order repaired and checked by reading, not compiled, not run on hardware
//
#include <hip/hip_runtime.h>
#include <hip/hip_bf16.h>

// ---------------------------------------------------------------------------
// MI455X (gfx1250) point-transformer block, wave32 + v_wmma_f32_16x16x32_f16.
//
// Round 3: removed predicated epilogues (R is always a multiple of 16; sf
// widened to 32 rows) so C-tile stores issue as straight-line clauses instead
// of 8-deep s_and_saveexec chains. Weights pre-packed to f16 (per-lane
// k-contiguous B-fragments -> two 16B global loads, no cvt in hot loop);
// A-fragments are two 16B LDS vector loads per ISA 7.12.2 layout. Fused
// per-point attention keeps all (B,N,K,DM) intermediates in LDS; flash-style
// 3-pass SA attention avoids materializing the 134 MB attention matrix.
// ---------------------------------------------------------------------------

#define B_   8
#define DP_  3
#define N_   2048
#define DM_  128
#define K_   20
#define C4_  32
#define BN_  (B_ * N_)

typedef __attribute__((ext_vector_type(16))) _Float16 v16h;
typedef __attribute__((ext_vector_type(8)))  _Float16 h8;
typedef __attribute__((ext_vector_type(8)))  float    v8f;

__device__ __forceinline__ v16h make_v16(h8 lo, h8 hi) {
  v16h a;
#pragma unroll
  for (int i = 0; i < 8; ++i) { a[i] = lo[i]; a[i + 8] = hi[i]; }
  return a;
}

// A (16x32 f16, ISA 7.12.2): lane -> M=lane&15; half=lane>>4 selects k runs
// {8h..8h+7} and {16+8h..16+8h+7}: two contiguous 16B chunks in a k-major row.
__device__ __forceinline__ v16h load_a_frag_lds(const _Float16* sh, int ld,
                                                int rowBase, int kBase, int lane) {
  int m = rowBase + (lane & 15), half = lane >> 4;
  const _Float16* p = sh + m * ld + kBase + 8 * half;
  h8 lo = *(const h8*)p;
  h8 hi = *(const h8*)(p + 16);
  return make_v16(lo, hi);
}

// B (32x16 f16): lane -> column N = lane&15; per-lane k runs {8h..8h+7} and
// {16+8h..16+8h+7} (matching A), in a packed f16 weight row (Wh is Cout x 128, row-major f16).
__device__ __forceinline__ v16h load_b_frag_wh(const _Float16* __restrict__ Wh, int ldw,
                                               int colBase, int kBase, int lane) {
  int c = colBase + (lane & 15), half = lane >> 4;
  const _Float16* p = Wh + (size_t)c * ldw + kBase + 8 * half;
  h8 lo = *(const h8*)p;
  h8 hi = *(const h8*)(p + 16);
  return make_v16(lo, hi);
}

#define WMMA_F16(acc, a, b) \
  __builtin_amdgcn_wmma_f32_16x16x32_f16(false, (a), false, (b), (short)0, (acc), false, false)

#define V8F_ZERO {0.f,0.f,0.f,0.f,0.f,0.f,0.f,0.f}

// ---- weight pre-conversion f32 -> f16 --------------------------------------
__global__ __launch_bounds__(256) void cvt_f16_kernel(const float* __restrict__ s,
                                                      _Float16* __restrict__ d, int n) {
  int i = blockIdx.x * 256 + threadIdx.x;
  if (i < n) d[i] = (_Float16)s[i];
}

// ---- kNN: per-point top-K of pd = 2*dot - |xi|^2 - |xm|^2 ------------------
__global__ __launch_bounds__(128) void knn_kernel(const float* __restrict__ features,
                                                  int* __restrict__ idx_out) {
  int tid = blockIdx.x * 128 + threadIdx.x;      // one thread per (b, n)
  if (tid >= BN_) return;
  int b = tid >> 11, n = tid & (N_ - 1);
  const float* fb = features + (size_t)b * DP_ * N_;
  float x0 = fb[n], x1 = fb[N_ + n], x2 = fb[2 * N_ + n];
  float xxn = x0 * x0 + x1 * x1 + x2 * x2;
  float bd[K_]; int bi[K_];
#pragma unroll
  for (int j = 0; j < K_; ++j) { bd[j] = -3.4e38f; bi[j] = 0; }
  for (int m = 0; m < N_; ++m) {
    float y0 = fb[m], y1 = fb[N_ + m], y2 = fb[2 * N_ + m];
    float pd = 2.f * (x0 * y0 + x1 * y1 + x2 * y2) - xxn - (y0 * y0 + y1 * y1 + y2 * y2);
    if (pd > bd[K_ - 1]) {
      int pos = K_ - 1;
#pragma unroll
      for (int j = K_ - 2; j >= 0; --j) if (pd > bd[j]) pos = j;
#pragma unroll
      for (int j = K_ - 1; j > 0; --j) if (j > pos) { bd[j] = bd[j - 1]; bi[j] = bi[j - 1]; }
#pragma unroll
      for (int j = 0; j < K_; ++j) if (j == pos) { bd[j] = pd; bi[j] = m; }
    }
  }
#pragma unroll
  for (int j = 0; j < K_; ++j) idx_out[(size_t)tid * K_ + j] = bi[j];
}

// ---- fc1: x1[bn,c] = features(b,:,n) . fc1_w[c,:] + b  (K-dim = 3, VALU) ---
__global__ __launch_bounds__(256) void fc1_kernel(const float* __restrict__ features,
                                                  const float* __restrict__ w,
                                                  const float* __restrict__ bias,
                                                  float* __restrict__ x1) {
  int i = blockIdx.x * 256 + threadIdx.x;
  if (i >= BN_ * DM_) return;
  int c = i & (DM_ - 1); int bn = i >> 7; int b = bn >> 11; int n = bn & (N_ - 1);
  const float* fb = features + (size_t)b * DP_ * N_;
  float s = bias[c];
#pragma unroll
  for (int d = 0; d < DP_; ++d) s += fb[d * N_ + n] * w[c * DP_ + d];
  x1[i] = s;
}

// ---- generic WMMA row-GEMM: Y(RxC) = [relu](X(Rx128) @ W(Cx128)^T + bias) --
// Requires R % 16 == 0 (always BN_ here): no row predication in the epilogue.
__global__ __launch_bounds__(256) void gemm_xwT_wmma_kernel(const float* __restrict__ X,
                                                            const _Float16* __restrict__ Wh,
                                                            const float* __restrict__ bias,
                                                            float* __restrict__ Y,
                                                            int C, int relu) {
  __shared__ _Float16 shA[16][136];
  int r0 = blockIdx.x * 16, t = threadIdx.x;
  // Warm WGP$/L2 with the packed f16 weights (global_prefetch_b8).
  if (t * 64 < C * DM_) __builtin_prefetch(Wh + (size_t)t * 64, 0, 0);
  {
    int m = t >> 4, k0 = (t & 15) * 8;           // 8 contiguous halves/thread
    const float* xp = X + (size_t)(r0 + m) * DM_ + k0;
    h8 v;
#pragma unroll
    for (int i = 0; i < 8; ++i) v[i] = (_Float16)xp[i];
    *(h8*)&shA[m][k0] = v;                       // one b128 ds_store
  }
  __syncthreads();
  int wave = t >> 5, lane = t & 31;
  for (int nt = wave; nt * 16 < C; nt += 8) {
    v8f acc = V8F_ZERO;
#pragma unroll
    for (int kk = 0; kk < 4; ++kk) {
      v16h a = load_a_frag_lds(&shA[0][0], 136, 0, kk * 32, lane);
      v16h bf = load_b_frag_wh(Wh, DM_, nt * 16, kk * 32, lane);
      acc = WMMA_F16(acc, a, bf);
    }
    int col = nt * 16 + (lane & 15), half = lane >> 4;
    float bv = bias ? bias[col] : 0.f;
    float* yp = Y + (size_t)(r0 + 8 * half) * C + col;
#pragma unroll
    for (int rr = 0; rr < 8; ++rr) {
      float v = acc[rr] + bv;
      if (relu) v = fmaxf(v, 0.f);
      yp[(size_t)rr * C] = v;
    }
  }
}

// ---- fused per-point attention: edge MLP, gamma MLP, log-softmax, reduce ---
__global__ __launch_bounds__(128) void point_attn_kernel(
    const float* __restrict__ features, const int* __restrict__ idx,
    const float* __restrict__ q, const float* __restrict__ kx, const float* __restrict__ vx,
    const float* __restrict__ fb1_w, const float* __restrict__ fb1_b,
    const _Float16* __restrict__ fb2_h, const float* __restrict__ fb2_b,
    const _Float16* __restrict__ fg1_h, const float* __restrict__ fg1_b,
    const _Float16* __restrict__ fg2_h, const float* __restrict__ fg2_b,
    float* __restrict__ attn_out, float* __restrict__ res_attn) {
  __shared__ int      sidx[K_];
  __shared__ float    se6[K_][8];
  __shared__ _Float16 shA[32][136];   // MLP activations (A-matrix source)
  __shared__ _Float16 shT[32][136];   // gamma hidden
  __shared__ _Float16 shKF[32][136];  // kf edge features
  __shared__ _Float16 shVG[32][136];  // gathered v
  __shared__ float    sf[32][DM_];    // pre-softmax logits (32 rows: no guard)

  int bn = blockIdx.x, b = bn >> 11, n = bn & (N_ - 1);
  int t = threadIdx.x, wave = t >> 5, lane = t & 31;

  for (int ii = t; ii < 32 * 136; ii += 128) {   // zero pad rows 20..31
    (&shA[0][0])[ii] = (_Float16)0.f; (&shT[0][0])[ii] = (_Float16)0.f;
  }
  if (t < K_) sidx[t] = idx[(size_t)bn * K_ + t];
  __syncthreads();

  if (t < K_) {                                   // edge features (nbr-ctr, ctr)
    int j = t, nb = sidx[j];
    const float* fb = features + (size_t)b * DP_ * N_;
#pragma unroll
    for (int d = 0; d < DP_; ++d) {
      float c0 = fb[d * N_ + n], nv = fb[d * N_ + nb];
      se6[j][d] = nv - c0; se6[j][DP_ + d] = c0;
    }
    se6[j][6] = 0.f; se6[j][7] = 0.f;
  }
  __syncthreads();

  {                                               // h1 = relu(e6 @ fb1^T + b), K-dim 6 (VALU)
    int c = t;
    float wc[6], bb = fb1_b[c];
#pragma unroll
    for (int d = 0; d < 6; ++d) wc[d] = fb1_w[c * 6 + d];
    for (int j = 0; j < K_; ++j) {
      float s = bb;
#pragma unroll
      for (int d = 0; d < 6; ++d) s += se6[j][d] * wc[d];
      shA[j][c] = (_Float16)fmaxf(s, 0.f);
    }
  }
  __syncthreads();

  // kf = h1 @ fb2^T + b   (WMMA: 2 M-tiles x 8 N-tiles over 4 waves)
  for (int tile = wave; tile < 16; tile += 4) {
    int mt = tile >> 3, nt = tile & 7;
    v8f acc = V8F_ZERO;
#pragma unroll
    for (int kk = 0; kk < 4; ++kk) {
      v16h a = load_a_frag_lds(&shA[0][0], 136, mt * 16, kk * 32, lane);
      v16h bf = load_b_frag_wh(fb2_h, DM_, nt * 16, kk * 32, lane);
      acc = WMMA_F16(acc, a, bf);
    }
    int col = nt * 16 + (lane & 15), half = lane >> 4;
    float bv = fb2_b[col];
#pragma unroll
    for (int rr = 0; rr < 8; ++rr) {
      int m = mt * 16 + rr + 8 * half;
      shKF[m][col] = (_Float16)(acc[rr] + bv);
    }
  }
  __syncthreads();

  {                                               // a = q - kg + kf ; cache vg
    int c = t;
    float qv = q[(size_t)bn * DM_ + c];
    for (int j = 0; j < K_; ++j) {
      int nb = sidx[j];
      size_t gi = ((size_t)b * N_ + nb) * DM_ + c;
      float kg = kx[gi], vg = vx[gi];
      shVG[j][c] = (_Float16)vg;
      shA[j][c] = (_Float16)(qv - kg + (float)shKF[j][c]);
    }
  }
  __syncthreads();

  // g1 = relu(a @ fg1^T + b)
  for (int tile = wave; tile < 16; tile += 4) {
    int mt = tile >> 3, nt = tile & 7;
    v8f acc = V8F_ZERO;
#pragma unroll
    for (int kk = 0; kk < 4; ++kk) {
      v16h a = load_a_frag_lds(&shA[0][0], 136, mt * 16, kk * 32, lane);
      v16h bf = load_b_frag_wh(fg1_h, DM_, nt * 16, kk * 32, lane);
      acc = WMMA_F16(acc, a, bf);
    }
    int col = nt * 16 + (lane & 15), half = lane >> 4;
    float bv = fg1_b[col];
#pragma unroll
    for (int rr = 0; rr < 8; ++rr) {
      int m = mt * 16 + rr + 8 * half;
      shT[m][col] = (_Float16)fmaxf(acc[rr] + bv, 0.f);
    }
  }
  __syncthreads();

  // a2 = g1 @ fg2^T + b  -> sf (f32, all 32 rows, unguarded)
  for (int tile = wave; tile < 16; tile += 4) {
    int mt = tile >> 3, nt = tile & 7;
    v8f acc = V8F_ZERO;
#pragma unroll
    for (int kk = 0; kk < 4; ++kk) {
      v16h a = load_a_frag_lds(&shT[0][0], 136, mt * 16, kk * 32, lane);
      v16h bf = load_b_frag_wh(fg2_h, DM_, nt * 16, kk * 32, lane);
      acc = WMMA_F16(acc, a, bf);
    }
    int col = nt * 16 + (lane & 15), half = lane >> 4;
    float bv = fg2_b[col];
#pragma unroll
    for (int rr = 0; rr < 8; ++rr) {
      int m = mt * 16 + rr + 8 * half;
      sf[m][col] = acc[rr] + bv;
    }
  }
  __syncthreads();

  {                                               // log-softmax over K, reduce
    int c = t;
    const float scale = 0.08838834764831845f;     // 1/sqrt(128)
    float v[K_], mx = -3.4e38f;
#pragma unroll
    for (int j = 0; j < K_; ++j) { v[j] = sf[j][c] * scale; mx = fmaxf(mx, v[j]); }
    float s = 0.f;
#pragma unroll
    for (int j = 0; j < K_; ++j) s += __expf(v[j] - mx);
    float lz = __logf(s) + mx;
    float resv = 0.f;
#pragma unroll
    for (int j = 0; j < K_; ++j) {
      float at = v[j] - lz;
      attn_out[((size_t)bn * K_ + j) * DM_ + c] = at;
      resv += at * ((float)shVG[j][c] + (float)shKF[j][c]);
    }
    res_attn[(size_t)bn * DM_ + c] = resv;
  }
}

// ---- BatchNorm stats: per-channel mean/var over B*N rows -------------------
__global__ __launch_bounds__(256) void bn_stats_kernel(const float* __restrict__ x,
                                                       float* __restrict__ stats, int rows) {
  __shared__ float ss[256], ss2[256];
  int c = blockIdx.x, t = threadIdx.x;
  float s = 0.f, s2 = 0.f;
  for (int r = t; r < rows; r += 256) {
    float v = x[(size_t)r * DM_ + c];
    s += v; s2 += v * v;
  }
  ss[t] = s; ss2[t] = s2; __syncthreads();
  for (int off = 128; off > 0; off >>= 1) {
    if (t < off) { ss[t] += ss[t + off]; ss2[t] += ss2[t + off]; }
    __syncthreads();
  }
  if (t == 0) {
    float mu = ss[0] / rows;
    stats[c] = mu;
    stats[DM_ + c] = ss2[0] / rows - mu * mu;
  }
}

// out = relu(bn(x)) + add
__global__ __launch_bounds__(256) void bn_apply_add_kernel(const float* __restrict__ x,
                                                           const float* __restrict__ stats,
                                                           const float* __restrict__ g,
                                                           const float* __restrict__ be,
                                                           const float* __restrict__ add,
                                                           float* __restrict__ out, int total) {
  int i = blockIdx.x * 256 + threadIdx.x;
  if (i >= total) return;
  int c = i & (DM_ - 1);
  float v = g[c] * (x[i] - stats[c]) * rsqrtf(stats[DM_ + c] + 1e-5f) + be[c];
  out[i] = fmaxf(v, 0.f) + add[i];
}

__global__ __launch_bounds__(256) void diff_kernel(const float* __restrict__ a,
                                                   const float* __restrict__ b,
                                                   float* __restrict__ o, int total) {
  int i = blockIdx.x * 256 + threadIdx.x;
  if (i < total) o[i] = a[i] - b[i];
}

// ---- SA flash pass 1: per-row max & sum of exp over 2048 columns -----------
__global__ __launch_bounds__(32) void sa_row_softmax_kernel(const float* __restrict__ xqk,
                                                            float* __restrict__ rmax,
                                                            float* __restrict__ rsum) {
  __shared__ float e[N_];
  int bn = blockIdx.x, b = bn >> 11, lane = threadIdx.x;
  float xn[C4_];
  const float* rq = xqk + (size_t)bn * C4_;
#pragma unroll
  for (int c = 0; c < C4_; ++c) xn[c] = rq[c];
  const float* base = xqk + (size_t)b * N_ * C4_;
  for (int m = lane; m < N_; m += 32) {
    const float* rm = base + (size_t)m * C4_;
    float d = 0.f;
#pragma unroll
    for (int c = 0; c < C4_; ++c) d += xn[c] * rm[c];
    e[m] = d;
  }
  __syncthreads();
  float mx = -3.4e38f;
  for (int m = lane; m < N_; m += 32) mx = fmaxf(mx, e[m]);
#pragma unroll
  for (int off = 16; off > 0; off >>= 1) mx = fmaxf(mx, __shfl_xor(mx, off, 32));
  float s = 0.f;
  for (int m = lane; m < N_; m += 32) s += __expf(e[m] - mx);
#pragma unroll
  for (int off = 16; off > 0; off >>= 1) s += __shfl_xor(s, off, 32);
  if (lane == 0) { rmax[bn] = mx; rsum[bn] = s; }
}

// ---- SA flash pass 2: column sums of the row-softmax (energy symmetric) ----
__global__ __launch_bounds__(32) void sa_colsum_kernel(const float* __restrict__ xqk,
                                                       const float* __restrict__ rmax,
                                                       const float* __restrict__ rsum,
                                                       float* __restrict__ colsum) {
  int bm = blockIdx.x, b = bm >> 11, lane = threadIdx.x;
  float xm[C4_];
  const float* rm_ = xqk + (size_t)bm * C4_;
#pragma unroll
  for (int c = 0; c < C4_; ++c) xm[c] = rm_[c];
  const float* base = xqk + (size_t)b * N_ * C4_;
  const float* rmb = rmax + (size_t)b * N_;
  const float* rsb = rsum + (size_t)b * N_;
  float s = 0.f;
  for (int n = lane; n < N_; n += 32) {
    const float* rn = base + (size_t)n * C4_;
    float d = 0.f;
#pragma unroll
    for (int c = 0; c < C4_; ++c) d += xm[c] * rn[c];
    s += __expf(d - rmb[n]) / rsb[n];
  }
#pragma unroll
  for (int off = 16; off > 0; off >>= 1) s += __shfl_xor(s, off, 32);
  if (lane == 0) colsum[bm] = s;
}

// ---- SA flash pass 3: xr[:, m] = sum_n xv[:, n] * p[n,m] / colsum[m] -------
__global__ __launch_bounds__(256) void sa_xr_kernel(const float* __restrict__ xqk,
                                                    const float* __restrict__ xv,
                                                    const float* __restrict__ rmax,
                                                    const float* __restrict__ rsum,
                                                    const float* __restrict__ colsum,
                                                    float* __restrict__ xr) {
  __shared__ float sm[16][C4_];
  __shared__ float sp[64][16];
  int b = blockIdx.y, mt = blockIdx.x, t = threadIdx.x;
  for (int ii = t; ii < 16 * C4_; ii += 256) {
    int mi = ii >> 5, c = ii & (C4_ - 1);
    sm[mi][c] = xqk[((size_t)b * N_ + mt * 16 + mi) * C4_ + c];
  }
  __syncthreads();
  float acc[8];
#pragma unroll
  for (int j = 0; j < 8; ++j) acc[j] = 0.f;
  int mi = t & 15, cg = t >> 4;
  for (int n0 = 0; n0 < N_; n0 += 64) {
    __syncthreads();
    for (int ii = t; ii < 64 * 16; ii += 256) {
      int nn = ii >> 4, mj = ii & 15; int n = n0 + nn;
      const float* rn = xqk + ((size_t)b * N_ + n) * C4_;
      float d = 0.f;
#pragma unroll
      for (int c = 0; c < C4_; ++c) d += rn[c] * sm[mj][c];
      sp[nn][mj] = __expf(d - rmax[b * N_ + n]) / rsum[b * N_ + n];
    }
    __syncthreads();
    for (int nn = 0; nn < 64; ++nn) {
      float pv = sp[nn][mi];
      const float* xvr = xv + ((size_t)b * N_ + n0 + nn) * DM_ + cg * 8;
#pragma unroll
      for (int j = 0; j < 8; ++j) acc[j] += xvr[j] * pv;
    }
  }
  int m = mt * 16 + mi;
  float inv = 1.f / (1e-9f + colsum[b * N_ + m]);
  float* o = xr + ((size_t)b * N_ + m) * DM_ + cg * 8;
#pragma unroll
  for (int j = 0; j < 8; ++j) o[j] = acc[j] * inv;
}

// ---------------------------------------------------------------------------
extern "C" void kernel_launch(void* const* d_in, const int* in_sizes, int n_in,
                              void* d_out, int out_size, void* d_ws, size_t ws_size,
                              hipStream_t stream) {
  const float* features = (const float*)d_in[0];
  const float* fc1_w = (const float*)d_in[1];  const float* fc1_b = (const float*)d_in[2];
  const float* fc2_w = (const float*)d_in[3];  const float* fc2_b = (const float*)d_in[4];
  const float* bn_g  = (const float*)d_in[5];  const float* bn_b  = (const float*)d_in[6];
  const float* fb1_w = (const float*)d_in[7];  const float* fb1_b = (const float*)d_in[8];
  const float* fb2_w = (const float*)d_in[9];  const float* fb2_b = (const float*)d_in[10];
  const float* fg1_w = (const float*)d_in[11]; const float* fg1_b = (const float*)d_in[12];
  const float* fg2_w = (const float*)d_in[13]; const float* fg2_b = (const float*)d_in[14];
  const float* wq = (const float*)d_in[15];
  const float* wk = (const float*)d_in[16];
  const float* wv = (const float*)d_in[17];
  const float* sa_qk_w = (const float*)d_in[18];
  const float* sa_v_w = (const float*)d_in[19]; const float* sa_v_b = (const float*)d_in[20];
  const float* sa_t_w = (const float*)d_in[21]; const float* sa_t_b = (const float*)d_in[22];
  const float* sa_bn_g = (const float*)d_in[23]; const float* sa_bn_b = (const float*)d_in[24];

  float* out = (float*)d_out;                       // (B,N,DM)
  float* attn_out = out + (size_t)BN_ * DM_;        // (B,N,K,DM)

  char* ws = (char*)d_ws; size_t off = 0;
  auto alloc = [&](size_t bytes) -> void* {
    void* p = ws + off; off = (off + bytes + 255) & ~(size_t)255; return p;
  };
  int*   idx     = (int*)  alloc((size_t)BN_ * K_ * 4);
  float* x1      = (float*)alloc((size_t)BN_ * DM_ * 4);
  float* qb      = (float*)alloc((size_t)BN_ * DM_ * 4);
  float* kxb     = (float*)alloc((size_t)BN_ * DM_ * 4);
  float* vxb     = (float*)alloc((size_t)BN_ * DM_ * 4);
  float* resat   = (float*)alloc((size_t)BN_ * DM_ * 4);
  float* res2    = (float*)alloc((size_t)BN_ * DM_ * 4);
  float* res     = (float*)alloc((size_t)BN_ * DM_ * 4);
  float* xqk     = (float*)alloc((size_t)BN_ * C4_ * 4);
  float* xv      = (float*)alloc((size_t)BN_ * DM_ * 4);
  float* rmax    = (float*)alloc((size_t)BN_ * 4);
  float* rsum    = (float*)alloc((size_t)BN_ * 4);
  float* csum    = (float*)alloc((size_t)BN_ * 4);
  float* xr      = (float*)alloc((size_t)BN_ * DM_ * 4);
  float* dif     = (float*)alloc((size_t)BN_ * DM_ * 4);
  float* xr2     = (float*)alloc((size_t)BN_ * DM_ * 4);
  float* st1     = (float*)alloc(2 * DM_ * 4);
  float* st2     = (float*)alloc(2 * DM_ * 4);
  // packed f16 weights (Cout x 128 row-major)
  const int WN = DM_ * DM_;                         // 16384
  _Float16* fb2_h  = (_Float16*)alloc((size_t)WN * 2);
  _Float16* fg1_h  = (_Float16*)alloc((size_t)WN * 2);
  _Float16* fg2_h  = (_Float16*)alloc((size_t)WN * 2);
  _Float16* wq_h   = (_Float16*)alloc((size_t)WN * 2);
  _Float16* wk_h   = (_Float16*)alloc((size_t)WN * 2);
  _Float16* wv_h   = (_Float16*)alloc((size_t)WN * 2);
  _Float16* fc2_h  = (_Float16*)alloc((size_t)WN * 2);
  _Float16* saqk_h = (_Float16*)alloc((size_t)C4_ * DM_ * 2);
  _Float16* sav_h  = (_Float16*)alloc((size_t)WN * 2);
  _Float16* sat_h  = (_Float16*)alloc((size_t)WN * 2);

  const int total = BN_ * DM_;
  const int rowTiles = BN_ / 16;                    // 1024
  const int cvg = (WN + 255) / 256;                 // 64

  cvt_f16_kernel<<<cvg, 256, 0, stream>>>(fb2_w, fb2_h, WN);
  cvt_f16_kernel<<<cvg, 256, 0, stream>>>(fg1_w, fg1_h, WN);
  cvt_f16_kernel<<<cvg, 256, 0, stream>>>(fg2_w, fg2_h, WN);
  cvt_f16_kernel<<<cvg, 256, 0, stream>>>(wq, wq_h, WN);
  cvt_f16_kernel<<<cvg, 256, 0, stream>>>(wk, wk_h, WN);
  cvt_f16_kernel<<<cvg, 256, 0, stream>>>(wv, wv_h, WN);
  cvt_f16_kernel<<<cvg, 256, 0, stream>>>(fc2_w, fc2_h, WN);
  cvt_f16_kernel<<<(C4_ * DM_ + 255) / 256, 256, 0, stream>>>(sa_qk_w, saqk_h, C4_ * DM_);
  cvt_f16_kernel<<<cvg, 256, 0, stream>>>(sa_v_w, sav_h, WN);
  cvt_f16_kernel<<<cvg, 256, 0, stream>>>(sa_t_w, sat_h, WN);

  knn_kernel<<<BN_ / 128, 128, 0, stream>>>(features, idx);
  fc1_kernel<<<(total + 255) / 256, 256, 0, stream>>>(features, fc1_w, fc1_b, x1);

  gemm_xwT_wmma_kernel<<<rowTiles, 256, 0, stream>>>(x1, wq_h, nullptr, qb,  DM_, 0);
  gemm_xwT_wmma_kernel<<<rowTiles, 256, 0, stream>>>(x1, wk_h, nullptr, kxb, DM_, 0);
  gemm_xwT_wmma_kernel<<<rowTiles, 256, 0, stream>>>(x1, wv_h, nullptr, vxb, DM_, 0);

  point_attn_kernel<<<BN_, 128, 0, stream>>>(features, idx, qb, kxb, vxb,
                                             fb1_w, fb1_b, fb2_h, fb2_b,
                                             fg1_h, fg1_b, fg2_h, fg2_b,
                                             attn_out, resat);

  gemm_xwT_wmma_kernel<<<rowTiles, 256, 0, stream>>>(resat, fc2_h, fc2_b, res2, DM_, 0);
  bn_stats_kernel<<<DM_, 256, 0, stream>>>(res2, st1, BN_);
  bn_apply_add_kernel<<<(total + 255) / 256, 256, 0, stream>>>(res2, st1, bn_g, bn_b, x1, res, total);

  gemm_xwT_wmma_kernel<<<rowTiles, 256, 0, stream>>>(res, saqk_h, nullptr, xqk, C4_, 0);
  gemm_xwT_wmma_kernel<<<rowTiles, 256, 0, stream>>>(res, sav_h, sa_v_b, xv, DM_, 0);

  sa_row_softmax_kernel<<<BN_, 32, 0, stream>>>(xqk, rmax, rsum);
  sa_colsum_kernel<<<BN_, 32, 0, stream>>>(xqk, rmax, rsum, csum);
  sa_xr_kernel<<<dim3(N_ / 16, B_), 256, 0, stream>>>(xqk, xv, rmax, rsum, csum, xr);

  diff_kernel<<<(total + 255) / 256, 256, 0, stream>>>(res, xr, dif, total);
  gemm_xwT_wmma_kernel<<<rowTiles, 256, 0, stream>>>(dif, sat_h, sa_t_b, xr2, DM_, 0);
  bn_stats_kernel<<<DM_, 256, 0, stream>>>(xr2, st2, BN_);
  bn_apply_add_kernel<<<(total + 255) / 256, 256, 0, stream>>>(xr2, st2, sa_bn_g, sa_bn_b, res, out, total);

  (void)in_sizes; (void)n_in; (void)out_size; (void)ws_size;
}
